// MLPDecoder_72378788873075
// MI455X (gfx1250) — matrix operand order repaired and checked by reading, not compiled, not run on hardware
//
#include <hip/hip_runtime.h>
#include <hip/hip_bf16.h>

typedef __attribute__((ext_vector_type(16))) __bf16 v16bf;
typedef __attribute__((ext_vector_type(8)))  __bf16 v8bf;
typedef __attribute__((ext_vector_type(8)))  float  v8f;
typedef __attribute__((ext_vector_type(4)))  float  v4f;
typedef __attribute__((ext_vector_type(4)))  unsigned int u32x4;
typedef __attribute__((ext_vector_type(8)))  int i32x8;
typedef __attribute__((ext_vector_type(4)))  int i32x4;

#define DIM    512
#define H1     128
#define H2     64
#define BATCH  2048
#define MTOT   (BATCH * 65)       // 133120 output rows (pos then neg, flat)
#define KC     32                 // K chunk = one bf16 WMMA K-step
#define WAVES  4                  // 4 waves / block, one 16-row M-tile each

// ---- CDNA5 TDM availability (guarded: host pass / older toolchains fall back) ----
#if defined(__has_builtin)
#  if __has_builtin(__builtin_amdgcn_tensor_load_to_lds) && __has_builtin(__builtin_amdgcn_s_wait_tensorcnt)
#    define HAS_TDM 1
#  endif
#endif
#ifndef HAS_TDM
#  define HAS_TDM 0
#endif
#if HAS_TDM
#  if __has_include(<hip/amd_detail/amd_gfx1250_TDM.h>)
#    define TDM_6ARG 1            // therock-10.0 headers -> 6-arg builtin
#  else
#    define TDM_6ARG 0            // ROCm 7.2 -> 5-arg builtin
#  endif
#endif

__device__ __forceinline__ v8bf cvt8(v4f a, v4f b) {
    v8bf r;
    r[0] = (__bf16)a[0]; r[1] = (__bf16)a[1]; r[2] = (__bf16)a[2]; r[3] = (__bf16)a[3];
    r[4] = (__bf16)b[0]; r[5] = (__bf16)b[1]; r[6] = (__bf16)b[2]; r[7] = (__bf16)b[3];
    return r;
}
__device__ __forceinline__ v16bf cat16(v8bf lo, v8bf hi) {
    return __builtin_shufflevector(lo, hi, 0,1,2,3,4,5,6,7,8,9,10,11,12,13,14,15);
}

// A fragment (16x32 bf16, ISA layout): row = lane&15, two 8-value K-runs
__device__ __forceinline__ v16bf build_afrag(const float* embL, const float* embR,
                                             int kc, int hh) {
    const float* pb = (kc < DIM) ? (embL + kc) : (embR + kc);   // uniform select
    const float* p0 = pb + 8 * hh;
    const float* p1 = pb + 16 + 8 * hh;
    return cat16(cvt8(*(const v4f*)p0, *(const v4f*)(p0 + 4)),
                 cvt8(*(const v4f*)p1, *(const v4f*)(p1 + 4)));
}

// 8 column tiles of layer-1 against one staged W1^T chunk
__device__ __forceinline__ void l1_step(const __bf16 (*sw)[40], v16bf afrag,
                                        v8f acc[8], int ln, int hh) {
    #pragma unroll
    for (int tt = 0; tt < 8; ++tt) {
        const __bf16* bp = &sw[tt * 16 + ln][8 * hh];
        v16bf bfrag = cat16(*(const v8bf*)bp, *(const v8bf*)(bp + 16));
        acc[tt] = __builtin_amdgcn_wmma_f32_16x16x32_bf16(
            false, afrag, false, bfrag, (short)0, acc[tt], false, false);
    }
}

#if HAS_TDM
// TDM: load W1^T tile [128 rows][32 cols] (2B elems, row stride 1024 elems)
// into LDS with 16B pad per 64B row -> 80B LDS rows (keeps b128 reads aligned)
__device__ __forceinline__ void tdm_load_w1(const __bf16* w1t, unsigned lds_off, int kc) {
    unsigned long long ga = (unsigned long long)(uintptr_t)w1t
                          + (unsigned long long)kc * 2ull;
    u32x4 g0 = { 1u,                                   // count=1 (valid), user mode
                 lds_off,                              // lds_addr (bytes)
                 (unsigned)ga,                         // global_addr[31:0]
                 ((unsigned)(ga >> 32) & 0x01FFFFFFu)  // global_addr[56:32]
                 | 0x80000000u };                      // type=2 ("image")
    i32x8 g1 = { (int)((1u << 16)                      // data_size = 2B
                     | (1u << 20)                      // pad_enable
                     | (3u << 22)                      // pad_interval = 16 dwords (64B)
                     | (3u << 25)),                    // pad_amount  = 4 dwords (16B)
                 (int)(1024u << 16),                   // tensor_dim0 = 1024 (lo16)
                 (int)(128u  << 16),                   // tensor_dim1 = 128  (lo16)
                 (int)(32u   << 16),                   // tile_dim0   = 32
                 128,                                  // tile_dim1=128, tile_dim2=0
                 1024,                                 // tensor_dim0_stride (lo32)
                 0, 0 };
    i32x4 gz = { 0, 0, 0, 0 };
#if TDM_6ARG
    i32x8 z8 = { 0, 0, 0, 0, 0, 0, 0, 0 };
    __builtin_amdgcn_tensor_load_to_lds(g0, g1, gz, gz, z8, 0);
#else
    __builtin_amdgcn_tensor_load_to_lds(g0, g1, gz, gz, 0);
#endif
}
#endif

// ---- prologue: one-shot bf16 transpose of W1 (1024x128) and W2 (128x64) ----
__global__ __launch_bounds__(256, 1)
void convert_weights_kernel(const float* __restrict__ W1, const float* __restrict__ W2,
                            __bf16* __restrict__ w1t, __bf16* __restrict__ w2t) {
    int idx = blockIdx.x * 256 + threadIdx.x;
    if (idx < 1024 * H1) {                 // coalesced read of W1 row-major
        int n = idx & 127, k = idx >> 7;
        w1t[n * 1024 + k] = (__bf16)W1[idx];
    } else {
        int j = idx - 1024 * H1;
        if (j < H1 * H2) {
            int n = j & 63, k = j >> 6;
            w2t[n * H1 + k] = (__bf16)W2[j];
        }
    }
}

// ======================= main kernel (TDM-staged weights) =======================
__global__ __launch_bounds__(128, 1)
void mlp_decoder_tdm_kernel(const float* __restrict__ embed,
                            const float* __restrict__ b1, const float* __restrict__ b2,
                            const float* __restrict__ W3, const float* __restrict__ b3,
                            const int*   __restrict__ hIdx, const int* __restrict__ tIdx,
                            const int*   __restrict__ n_s,
                            const __bf16* __restrict__ w1t, const __bf16* __restrict__ w2t,
                            float* __restrict__ out)
{
    __shared__ __align__(16) __bf16 sWt[2][H1][40];       // 20480 B (double buffer)
    __shared__ __align__(16) __bf16 sW2t[H2 * H1];        // 16384 B  [64][128]
    __shared__ __align__(16) __bf16 sZ[WAVES][16][136];   // 17408 B

    const int tid  = threadIdx.x;
    const int lane = tid & 31;
    const int w    = tid >> 5;
    const int hh   = lane >> 4;
    const int ln   = lane & 15;

#if HAS_TDM
    const unsigned lds0 = (unsigned)(uintptr_t)&sWt[0][0][0];
    const unsigned lds1 = (unsigned)(uintptr_t)&sWt[1][0][0];
    if (w == 0) tdm_load_w1(w1t, lds0, 0);        // kick first chunk via TDM
#endif

    // stage W2^T (already bf16) into LDS: 8 x (128 lanes x 16B)
    #pragma unroll
    for (int i = 0; i < 8; ++i) {
        int flat = i * 1024 + tid * 8;
        *(v8bf*)&sW2t[flat] = *(const v8bf*)&w2t[flat];
    }

    // ---- decode this lane's row -> (left,right) embedding pair ----
    const int mt = blockIdx.x * WAVES + w;
    const int m  = mt * 16 + ln;
    int li, ri;
    if (m < BATCH) {                               // positive rows
        li = hIdx[m]; ri = tIdx[m];
    } else {                                       // negative rows
        int idx = m - BATCH;
        int b_ = idx >> 6, j = idx & 63;
        int ns = n_s[(b_ << 6) + j];
        if (j < 32) { li = hIdx[b_]; ri = ns;       }
        else        { li = ns;       ri = tIdx[b_]; }
    }
    const float* embL = embed + (long)li * DIM;
    const float* embR = embed + (long)ri * DIM - DIM;

    float b1c[8], b2c[4], w3c[4];
    #pragma unroll
    for (int tt = 0; tt < 8; ++tt) b1c[tt] = b1[tt * 16 + ln];
    #pragma unroll
    for (int tt = 0; tt < 4; ++tt) { b2c[tt] = b2[tt * 16 + ln]; w3c[tt] = W3[tt * 16 + ln]; }
    const float b3v = b3[0];

    v8f acc[8];
    #pragma unroll
    for (int tt = 0; tt < 8; ++tt)
        #pragma unroll
        for (int v = 0; v < 8; ++v) acc[tt][v] = 0.0f;

    // ================= Layer 1: [16 x 1024] @ [1024 x 128] =================
#if HAS_TDM
    if (w == 0) __builtin_amdgcn_s_wait_tensorcnt(0);
    __syncthreads();
    int buf = 0;
    for (int kc = 0; kc < 2 * DIM; kc += KC) {
        if (w == 0 && kc + KC < 2 * DIM)
            tdm_load_w1(w1t, buf ? lds0 : lds1, kc + KC);   // prefetch next chunk
        v16bf afrag = build_afrag(embL, embR, kc, hh);
        l1_step(sWt[buf], afrag, acc, ln, hh);
        __syncthreads();                                    // all reads of buf done
        if (w == 0) __builtin_amdgcn_s_wait_tensorcnt(0);   // next buffer landed
        __syncthreads();
        buf ^= 1;
    }
#else
    __syncthreads();
    for (int kc = 0; kc < 2 * DIM; kc += KC) {
        {   // cooperative copy: thread tid = column n, 32 bf16 (4 x v8bf)
            const __bf16* src = w1t + tid * 1024 + kc;
            #pragma unroll
            for (int q = 0; q < 4; ++q)
                *(v8bf*)&sWt[0][tid][q * 8] = *(const v8bf*)(src + q * 8);
        }
        __syncthreads();
        v16bf afrag = build_afrag(embL, embR, kc, hh);
        l1_step(sWt[0], afrag, acc, ln, hh);
        __syncthreads();
    }
#endif

    // ---- bias + ReLU, write z1 tile (bf16) to per-wave LDS ----
    #pragma unroll
    for (int tt = 0; tt < 8; ++tt)
        #pragma unroll
        for (int v = 0; v < 8; ++v) {
            float z = acc[tt][v] + b1c[tt];
            z = z > 0.0f ? z : 0.0f;
            sZ[w][v + 8 * hh][tt * 16 + ln] = (__bf16)z;
        }

    // ================= Layer 2: [16 x 128] @ [128 x 64] =================
    v8f acc2[4];
    #pragma unroll
    for (int tt = 0; tt < 4; ++tt)
        #pragma unroll
        for (int v = 0; v < 8; ++v) acc2[tt][v] = 0.0f;

    #pragma unroll
    for (int k2 = 0; k2 < H1; k2 += 32) {
        const __bf16* ap = &sZ[w][ln][k2 + 8 * hh];
        v16bf afrag2 = cat16(*(const v8bf*)ap, *(const v8bf*)(ap + 16));
        #pragma unroll
        for (int tt = 0; tt < 4; ++tt) {
            const __bf16* bp = &sW2t[(tt * 16 + ln) * H1 + k2 + 8 * hh];
            v16bf bfrag = cat16(*(const v8bf*)bp, *(const v8bf*)(bp + 16));
            acc2[tt] = __builtin_amdgcn_wmma_f32_16x16x32_bf16(
                false, afrag2, false, bfrag, (short)0, acc2[tt], false, false);
        }
    }

    // ================= Layer 3: bias + ReLU, dot W3, + b3 =================
    float p[8];
    #pragma unroll
    for (int v = 0; v < 8; ++v) p[v] = 0.0f;
    #pragma unroll
    for (int tt = 0; tt < 4; ++tt)
        #pragma unroll
        for (int v = 0; v < 8; ++v) {
            float z = acc2[tt][v] + b2c[tt];
            z = z > 0.0f ? z : 0.0f;
            p[v] += z * w3c[tt];
        }

    float* zs2 = (float*)&sZ[w][0][0];
    #pragma unroll
    for (int v = 0; v < 8; ++v) zs2[(v + 8 * hh) * 16 + ln] = p[v];

    if (lane < 16) {
        float s = b3v;
        const float* rp = zs2 + lane * 16;
        #pragma unroll
        for (int i = 0; i < 16; ++i) s += rp[i];
        out[mt * 16 + lane] = s;
    }
}

// ============== fallback (no workspace): round-1 self-contained kernel ==============
__global__ __launch_bounds__(128, 1)
void mlp_decoder_fallback_kernel(const float* __restrict__ embed,
                        const float* __restrict__ W1, const float* __restrict__ b1,
                        const float* __restrict__ W2, const float* __restrict__ b2,
                        const float* __restrict__ W3, const float* __restrict__ b3,
                        const int*   __restrict__ hIdx, const int* __restrict__ tIdx,
                        const int*   __restrict__ n_s,
                        float* __restrict__ out)
{
    __shared__ __align__(16) __bf16 sWt[H1][40];
    __shared__ __align__(16) __bf16 sW2t[H2][136];
    __shared__ __align__(16) __bf16 sZ[WAVES][16][136];

    const int tid  = threadIdx.x;
    const int lane = tid & 31;
    const int w    = tid >> 5;
    const int hh   = lane >> 4;
    const int ln   = lane & 15;

    #pragma unroll
    for (int i = 0; i < (H1 * H2) / 128; ++i) {
        int pos = i * 128 + tid;
        int k = pos >> 6, n = pos & 63;
        sW2t[n][k] = (__bf16)W2[pos];
    }
    __syncthreads();

    const int mt = blockIdx.x * WAVES + w;
    const int m  = mt * 16 + ln;
    int li, ri;
    if (m < BATCH) { li = hIdx[m]; ri = tIdx[m]; }
    else {
        int idx = m - BATCH;
        int b_ = idx >> 6, j = idx & 63;
        int ns = n_s[(b_ << 6) + j];
        if (j < 32) { li = hIdx[b_]; ri = ns;       }
        else        { li = ns;       ri = tIdx[b_]; }
    }
    const float* embL = embed + (long)li * DIM;
    const float* embR = embed + (long)ri * DIM - DIM;

    float b1c[8], b2c[4], w3c[4];
    #pragma unroll
    for (int tt = 0; tt < 8; ++tt) b1c[tt] = b1[tt * 16 + ln];
    #pragma unroll
    for (int tt = 0; tt < 4; ++tt) { b2c[tt] = b2[tt * 16 + ln]; w3c[tt] = W3[tt * 16 + ln]; }
    const float b3v = b3[0];

    v8f acc[8];
    #pragma unroll
    for (int tt = 0; tt < 8; ++tt)
        #pragma unroll
        for (int v = 0; v < 8; ++v) acc[tt][v] = 0.0f;

    for (int kc = 0; kc < 2 * DIM; kc += KC) {
        {
            const int k  = tid >> 2;
            const int cg = (tid & 3) * 32;
            const float* wp = W1 + (kc + k) * H1 + cg;
            if (kc + KC < 2 * DIM) __builtin_prefetch(wp + KC * H1, 0, 3);
            #pragma unroll
            for (int q = 0; q < 8; ++q) {
                v4f v = *(const v4f*)(wp + 4 * q);
                #pragma unroll
                for (int e = 0; e < 4; ++e) sWt[cg + 4 * q + e][k] = (__bf16)v[e];
            }
        }
        __syncthreads();
        v16bf afrag = build_afrag(embL, embR, kc, hh);
        l1_step(sWt, afrag, acc, ln, hh);
        __syncthreads();
    }

    #pragma unroll
    for (int tt = 0; tt < 8; ++tt)
        #pragma unroll
        for (int v = 0; v < 8; ++v) {
            float z = acc[tt][v] + b1c[tt];
            z = z > 0.0f ? z : 0.0f;
            sZ[w][v + 8 * hh][tt * 16 + ln] = (__bf16)z;
        }

    v8f acc2[4];
    #pragma unroll
    for (int tt = 0; tt < 4; ++tt)
        #pragma unroll
        for (int v = 0; v < 8; ++v) acc2[tt][v] = 0.0f;

    #pragma unroll
    for (int k2 = 0; k2 < H1; k2 += 32) {
        const __bf16* ap = &sZ[w][ln][k2 + 8 * hh];
        v16bf afrag2 = cat16(*(const v8bf*)ap, *(const v8bf*)(ap + 16));
        #pragma unroll
        for (int tt = 0; tt < 4; ++tt) {
            const __bf16* bp = &sW2t[tt * 16 + ln][k2 + 8 * hh];
            v16bf bfrag = cat16(*(const v8bf*)bp, *(const v8bf*)(bp + 16));
            acc2[tt] = __builtin_amdgcn_wmma_f32_16x16x32_bf16(
                false, afrag2, false, bfrag, (short)0, acc2[tt], false, false);
        }
    }

    float p[8];
    #pragma unroll
    for (int v = 0; v < 8; ++v) p[v] = 0.0f;
    #pragma unroll
    for (int tt = 0; tt < 4; ++tt)
        #pragma unroll
        for (int v = 0; v < 8; ++v) {
            float z = acc2[tt][v] + b2c[tt];
            z = z > 0.0f ? z : 0.0f;
            p[v] += z * w3c[tt];
        }

    float* zs2 = (float*)&sZ[w][0][0];
    #pragma unroll
    for (int v = 0; v < 8; ++v) zs2[(v + 8 * hh) * 16 + ln] = p[v];

    if (lane < 16) {
        float s = b3v;
        const float* rp = zs2 + lane * 16;
        #pragma unroll
        for (int i = 0; i < 16; ++i) s += rp[i];
        out[mt * 16 + lane] = s;
    }
}

extern "C" void kernel_launch(void* const* d_in, const int* in_sizes, int n_in,
                              void* d_out, int out_size, void* d_ws, size_t ws_size,
                              hipStream_t stream) {
    const float* embed = (const float*)d_in[0];
    const float* W1    = (const float*)d_in[1];
    const float* b1    = (const float*)d_in[2];
    const float* W2    = (const float*)d_in[3];
    const float* b2    = (const float*)d_in[4];
    const float* W3    = (const float*)d_in[5];
    const float* b3    = (const float*)d_in[6];
    const int*   h     = (const int*)d_in[7];
    const int*   t     = (const int*)d_in[8];
    const int*   n_s   = (const int*)d_in[9];
    float* out = (float*)d_out;

    const int blocks = (MTOT / 16) / WAVES;                     // 2080
    const size_t need = (size_t)(1024 * H1 + H1 * H2) * 2;      // 278528 B

    if (ws_size >= need) {
        __bf16* w1t = (__bf16*)d_ws;
        __bf16* w2t = w1t + 1024 * H1;
        convert_weights_kernel<<<(1024 * H1 + H1 * H2 + 255) / 256, 256, 0, stream>>>(
            W1, W2, w1t, w2t);
        mlp_decoder_tdm_kernel<<<blocks, 128, 0, stream>>>(
            embed, b1, b2, W3, b3, h, t, n_s, w1t, w2t, out);
    } else {
        mlp_decoder_fallback_kernel<<<blocks, 128, 0, stream>>>(
            embed, W1, b1, W2, b2, W3, b3, h, t, n_s, out);
    }
}
